// TransformAttentionModel_41257455845611
// MI455X (gfx1250) — matrix operand order repaired and checked by reading, not compiled, not run on hardware
//
#include <hip/hip_runtime.h>

// ---------------------------------------------------------------------------
// Fused TransformAttention for MI455X (gfx1250, wave32, WMMA bf16 16x16x32)
// B=4 T=64 N=1024 H=128 Td=64 heads=8 hd=16
// Grid: 512 WGs x 256 threads (8 waves). WG owns one b and 8 nodes.
// ---------------------------------------------------------------------------

typedef __bf16 bf16;
typedef __attribute__((ext_vector_type(8)))  __bf16 v8bf;
typedef __attribute__((ext_vector_type(16))) __bf16 v16bf;
typedef __attribute__((ext_vector_type(8)))  float  v8f;

// ---- LDS layout (bytes); all region bases & row strides multiples of 16 ----
#define LDS_STEQ 0        // 64 rows x 144B (64 bf16 + pad)
#define LDS_STEP 9216     // 64 x 144B
#define LDS_X    18432    // 64 x 272B (128 bf16 + pad)
#define LDS_QO   35840    // 64 x 272B   Q, later overwritten by O (per-head cols)
#define LDS_K    53248    // 64 x 272B
#define LDS_VT   70656    // 128 x 144B  V transposed: [hd-col][time]
#define LDS_SCR  89088    // 8 waves x (64 x 80B) softmax(S) scratch
#define LDS_WQ   130048   // 128 x 400B  W transposed [col][k], k=0..191 + pad
#define LDS_WK   181248
#define LDS_WV   232448
#define LDS_TOTAL 283648  // ~277 KB of the 320 KB WGP LDS

static __device__ __forceinline__ v16bf cat8(v8bf lo, v8bf hi) {
  return __builtin_shufflevector(lo, hi, 0,1,2,3,4,5,6,7,8,9,10,11,12,13,14,15);
}
static __device__ __forceinline__ v8bf zero8bf() {
  v8bf z;
#pragma unroll
  for (int i = 0; i < 8; ++i) z[i] = (bf16)0.0f;
  return z;
}
static __device__ __forceinline__ v8f zero8f() {
  v8f z;
#pragma unroll
  for (int i = 0; i < 8; ++i) z[i] = 0.0f;
  return z;
}
// A fragment (16x32 bf16), full K=32, from row-major LDS.
// Lane half hl=0 holds K {koff+0..15}; hl=1 holds {+16..31}.
static __device__ __forceinline__ v16bf ldA32(const char* p, int strideB, int row, int koffB, int hl) {
  const char* q = p + row * strideB + koffB + hl * 32;
  return cat8(*(const v8bf*)q, *(const v8bf*)(q + 16));
}
// A fragment with only K=0..15 valid (upper lane half supplies zeros)
static __device__ __forceinline__ v16bf ldA16(const char* p, int strideB, int row, int koffB, int hl) {
  if (hl) return cat8(zero8bf(), zero8bf());
  const char* q = p + row * strideB + koffB;
  return cat8(*(const v8bf*)q, *(const v8bf*)(q + 16));
}
// B fragment (32x16 bf16): lane holds column `row`; 16 consecutive K per lane half.
static __device__ __forceinline__ v16bf ldB32(const char* p, int strideB, int row, int koffB, int hl) {
  const char* q = p + row * strideB + koffB + hl * 32;
  return cat8(*(const v8bf*)q, *(const v8bf*)(q + 16));
}
// B fragment with only K=0..15 valid (upper lane half supplies zeros)
static __device__ __forceinline__ v16bf ldB16(const char* p, int strideB, int row, int koffB, int hl) {
  if (hl) return cat8(zero8bf(), zero8bf());
  const char* q = p + row * strideB + koffB;
  return cat8(*(const v8bf*)q, *(const v8bf*)(q + 16));
}

#define WMMA(a, b, c) \
  __builtin_amdgcn_wmma_f32_16x16x32_bf16(false, (a), false, (b), (short)0, (c), false, false)

__global__ __launch_bounds__(256, 1)
void ta_fused_kernel(const float* __restrict__ X,
                     const float* __restrict__ STE_P, const float* __restrict__ STE_Q,
                     const float* __restrict__ Wq, const float* __restrict__ bq,
                     const float* __restrict__ Wk, const float* __restrict__ bk,
                     const float* __restrict__ Wv, const float* __restrict__ bv,
                     const float* __restrict__ Wc, const float* __restrict__ bc,
                     float* __restrict__ out) {
  extern __shared__ char smem[];
  const int tid  = threadIdx.x;
  const int wave = tid >> 5;       // 0..7: column strip / head id
  const int lane = tid & 31;
  const int lrow = lane & 15;      // M (A), N (B/C) index within tile
  const int hl   = lane >> 4;      // lane half
  const int cs   = wave * 16;      // owned column strip base
  const int b    = blockIdx.x >> 7;        // / 128
  const int n0   = (blockIdx.x & 127) * 8; // 8 nodes per WG

  // -------- one-time per WG: stage weights (bf16, transposed) + STE ---------
  {
    const float* Ws[3] = {Wq, Wk, Wv};
    char* Ls[3] = {smem + LDS_WQ, smem + LDS_WK, smem + LDS_WV};
    for (int m = 0; m < 3; ++m) {
      const float* W = Ws[m];
      char* L = Ls[m];
      for (int i = tid; i < 192 * 128; i += 256) {
        int k = i >> 7, c = i & 127;
        *(bf16*)(L + c * 400 + k * 2) = (bf16)W[i];
      }
    }
    for (int i = tid; i < 64 * 64; i += 256) {
      int t = i >> 6, d = i & 63;
      *(bf16*)(smem + LDS_STEQ + t * 144 + d * 2) = (bf16)STE_Q[(b * 64 + t) * 64 + d];
      *(bf16*)(smem + LDS_STEP + t * 144 + d * 2) = (bf16)STE_P[(b * 64 + t) * 64 + d];
    }
  }
  // W_comb: 8 persistent B-fragments in VGPRs (64 regs/lane), saves 64KB LDS
  v16bf wc[2][4];
#pragma unroll
  for (int i = 0; i < 2; ++i) {
#pragma unroll
    for (int kb = 0; kb < 4; ++kb) {
#pragma unroll
      for (int e = 0; e < 8; ++e) {
        int k = kb * 32 + hl * 16 + 2 * e;
        wc[i][kb][2 * e]     = (bf16)Wc[(i * 128 + k) * 128 + cs + lrow];
        wc[i][kb][2 * e + 1] = (bf16)Wc[(i * 128 + k + 1) * 128 + cs + lrow];
      }
    }
  }
  const float biasq = bq[cs + lrow], biask = bk[cs + lrow];
  const float biasv = bv[cs + lrow], biasc = bc[cs + lrow];
  __syncthreads();

#pragma unroll 1
  for (int nn = 0; nn < 8; ++nn) {
    const int n = n0 + nn;
    // ---- stage X[b,:,n,:] tile as bf16 ----
    for (int i = tid; i < 64 * 128; i += 256) {
      int t = i >> 7, c = i & 127;
      float xv = X[(((size_t)b * 64 + t) * 1024 + n) * 128 + c];
      *(bf16*)(smem + LDS_X + t * 272 + c * 2) = (bf16)xv;
    }
    if (nn + 1 < 8 && tid < 64)  // gfx1250 global_prefetch_b8 for next node's X
      __builtin_prefetch(&X[(((size_t)b * 64 + tid) * 1024 + (n + 1)) * 128], 0, 0);
    __syncthreads();

    // ================= QKV projections: relu([STE|X]@W + bias) =============
#pragma unroll
    for (int rb = 0; rb < 4; ++rb) {
      v8f aq = zero8f(), ak = zero8f(), av = zero8f();
#pragma unroll
      for (int kb = 0; kb < 6; ++kb) {
        const int row = 16 * rb + lrow;
        v16bf aKV, aQf;
        if (kb < 2) {  // K-slab from STE (64 wide)
          aKV = ldA32(smem + LDS_STEP, 144, row, kb * 64, hl);
          aQf = ldA32(smem + LDS_STEQ, 144, row, kb * 64, hl);
        } else {       // K-slab from X (128 wide)
          aKV = ldA32(smem + LDS_X, 272, row, (kb - 2) * 64, hl);
          aQf = aKV;
        }
        v16bf bQ = ldB32(smem + LDS_WQ, 400, cs + lrow, kb * 64, hl);
        v16bf bK = ldB32(smem + LDS_WK, 400, cs + lrow, kb * 64, hl);
        v16bf bV = ldB32(smem + LDS_WV, 400, cs + lrow, kb * 64, hl);
        aq = WMMA(aQf, bQ, aq);
        ak = WMMA(aKV, bK, ak);
        av = WMMA(aKV, bV, av);
      }
#pragma unroll
      for (int j = 0; j < 8; ++j) {  // C layout: row = 16rb + 8*hl + j, col = cs+lrow
        int trow = 16 * rb + 8 * hl + j;
        float fq = fmaxf(aq[j] + biasq, 0.f);
        float fk = fmaxf(ak[j] + biask, 0.f);
        float fv = fmaxf(av[j] + biasv, 0.f);
        *(bf16*)(smem + LDS_QO + trow * 272 + (cs + lrow) * 2) = (bf16)fq;
        *(bf16*)(smem + LDS_K  + trow * 272 + (cs + lrow) * 2) = (bf16)fk;
        *(bf16*)(smem + LDS_VT + (cs + lrow) * 144 + trow * 2) = (bf16)fv;  // transposed
      }
    }
    __syncthreads();

    // ============ attention, wave == head h; softmax over q axis ============
    const int h = wave;
    char* scr = smem + LDS_SCR + wave * 5120;  // per-wave [64 q][32 p] bf16, stride 80B

    v8f oacc[4];
#pragma unroll
    for (int qi = 0; qi < 4; ++qi) oacc[qi] = zero8f();

#pragma unroll
    for (int pjj = 0; pjj < 2; ++pjj) {  // two 32-wide p blocks
#pragma unroll
      for (int p2 = 0; p2 < 2; ++p2) {   // 16-wide p column tiles
        const int pblk = pjj * 2 + p2;
        v16bf bK = ldB16(smem + LDS_K, 272, 16 * pblk + lrow, h * 32, hl);
        v8f s[4];
#pragma unroll
        for (int qi = 0; qi < 4; ++qi) {
          // Q_h A-fragment reloaded per use (cheap LDS) to keep VGPRs <= 256
          v16bf aQ = ldA16(smem + LDS_QO, 272, 16 * qi + lrow, h * 32, hl);
          s[qi] = WMMA(aQ, bK, zero8f());
        }
        // softmax over q: this lane's column p = 16*pblk + lrow; q values are
        // {16qi + 8hl + j}; partner lane (xor 16) holds the other 32 q's.
        float mx = -1e30f;
#pragma unroll
        for (int qi = 0; qi < 4; ++qi)
#pragma unroll
          for (int j = 0; j < 8; ++j) mx = fmaxf(mx, s[qi][j]);
        mx = fmaxf(mx, __shfl_xor(mx, 16, 32));
        float sum = 0.f;
#pragma unroll
        for (int qi = 0; qi < 4; ++qi)
#pragma unroll
          for (int j = 0; j < 8; ++j) {
            float e = __expf(s[qi][j] - mx);
            s[qi][j] = e;
            sum += e;
          }
        sum += __shfl_xor(sum, 16, 32);
        float inv = 1.0f / sum;
#pragma unroll
        for (int qi = 0; qi < 4; ++qi)
#pragma unroll
          for (int j = 0; j < 8; ++j)
            *(bf16*)(scr + (16 * qi + 8 * hl + j) * 80 + (p2 * 16 + lrow) * 2) =
                (bf16)(s[qi][j] * inv);
      }
      // O += S[:, 32p block] @ V_h  (A from scratch, B from transposed V)
      v16bf bV = ldB32(smem + LDS_VT, 144, h * 16 + lrow, pjj * 64, hl);
#pragma unroll
      for (int qi = 0; qi < 4; ++qi) {
        v16bf aS = ldA32(scr, 80, 16 * qi + lrow, 0, hl);
        oacc[qi] = WMMA(aS, bV, oacc[qi]);
      }
    }
    // overwrite this head's Q columns with O (only this wave ever read them)
#pragma unroll
    for (int qi = 0; qi < 4; ++qi)
#pragma unroll
      for (int j = 0; j < 8; ++j)
        *(bf16*)(smem + LDS_QO + (16 * qi + 8 * hl + j) * 272 + (h * 16 + lrow) * 2) =
            (bf16)oacc[qi][j];
    __syncthreads();

    // ============ final combine: Y = X@Wc0 + O@Wc1 + b_comb, store f32 ======
#pragma unroll
    for (int rb = 0; rb < 4; ++rb) {
      v8f acc = zero8f();
#pragma unroll
      for (int kb = 0; kb < 4; ++kb) {
        v16bf aX = ldA32(smem + LDS_X,  272, 16 * rb + lrow, kb * 64, hl);
        v16bf aO = ldA32(smem + LDS_QO, 272, 16 * rb + lrow, kb * 64, hl);
        acc = WMMA(aX, wc[0][kb], acc);
        acc = WMMA(aO, wc[1][kb], acc);
      }
#pragma unroll
      for (int j = 0; j < 8; ++j) {
        int t = 16 * rb + 8 * hl + j;
        out[(((size_t)b * 64 + t) * 1024 + n) * 128 + cs + lrow] = acc[j] + biasc;
      }
    }
    __syncthreads();  // before next node's staging overwrites sX/sQ/sK/sV
  }
}

extern "C" void kernel_launch(void* const* d_in, const int* in_sizes, int n_in,
                              void* d_out, int out_size, void* d_ws, size_t ws_size,
                              hipStream_t stream) {
  const float* X     = (const float*)d_in[0];
  const float* STE_P = (const float*)d_in[1];
  const float* STE_Q = (const float*)d_in[2];
  const float* Wq    = (const float*)d_in[3];
  const float* bq    = (const float*)d_in[4];
  const float* Wk    = (const float*)d_in[5];
  const float* bk    = (const float*)d_in[6];
  const float* Wv    = (const float*)d_in[7];
  const float* bv    = (const float*)d_in[8];
  const float* Wc    = (const float*)d_in[9];
  const float* bc    = (const float*)d_in[10];
  float* out = (float*)d_out;

  (void)hipFuncSetAttribute((const void*)ta_fused_kernel,
                            hipFuncAttributeMaxDynamicSharedMemorySize, LDS_TOTAL);
  ta_fused_kernel<<<dim3(512), dim3(256), LDS_TOTAL, stream>>>(
      X, STE_P, STE_Q, Wq, bq, Wk, bk, Wv, bv, Wc, bc, out);
}
